// LBLHighwayBiLm_71021579206697
// MI455X (gfx1250) — matrix operand order repaired and checked by reading, not compiled, not run on hardware
//
#include <hip/hip_runtime.h>
#include <cstdint>
#include <cstddef>

// Problem constants (match reference)
#define B_    32
#define L_    1024
#define D_    512
#define W_    4
#define NL_   2
#define M_    (B_ * L_)     // 32768 rows
#define K_    D_            // 512
#define TWOD_ (2 * D_)      // 1024

// GEMM tiling
#define BM 128
#define BN 64     // nonlinear columns per workgroup (gate columns mirrored at +D)
#define BK 32
#define SA 40     // LDS row stride (ushorts), padded vs bank conflicts (16B aligned)
#define SB 40

typedef __attribute__((ext_vector_type(16))) __bf16         v16bf;
typedef __attribute__((ext_vector_type(8)))  float          v8f;
typedef __attribute__((ext_vector_type(4)))  float          f32x4;
typedef __attribute__((ext_vector_type(4)))  unsigned int   u32x4;
typedef __attribute__((ext_vector_type(4)))  unsigned short u16x4;
typedef __attribute__((ext_vector_type(8)))  unsigned short u16x8;

// Types matching the async-load builtin's parameters exactly:
// (int __vector(4) addrspace(1)*, int __vector(4) addrspace(3)*, Ii, Ii)
typedef int v4i_n __attribute__((vector_size(16)));
typedef __attribute__((address_space(1))) v4i_n gv4i;
typedef __attribute__((address_space(3))) v4i_n lv4i;

#if __has_builtin(__builtin_amdgcn_global_load_async_to_lds_b128)
#define ASYNC_COPY 1
#else
#define ASYNC_COPY 0
#endif

static __device__ __forceinline__ unsigned short f32_to_bf16_rne(float f) {
  unsigned int u = __float_as_uint(f);
  u += 0x7FFFu + ((u >> 16) & 1u);   // round-to-nearest-even
  return (unsigned short)(u >> 16);
}

// 16-byte global -> LDS copy: async (ASYNCcnt-tracked) if available.
static __device__ __forceinline__ void copy16_g2l(const unsigned short* g,
                                                  unsigned short* l) {
#if ASYNC_COPY
  __builtin_amdgcn_global_load_async_to_lds_b128(
      (gv4i*)(uintptr_t)g,
      (lv4i*)(unsigned)(uintptr_t)l,
      0, 0);
#else
  *(u16x8*)l = *(const u16x8*)g;
#endif
}

static __device__ __forceinline__ void wait_async_copies() {
#if ASYNC_COPY
#if __has_builtin(__builtin_amdgcn_s_wait_asynccnt)
  __builtin_amdgcn_s_wait_asynccnt(0);
#else
  asm volatile("s_wait_asynccnt 0" ::: "memory");
#endif
#endif
}

// ---------------------------------------------------------------------------
// Weights: f32 [K][2D] row-major -> bf16 [2D][K] (n-major / transposed).
// Makes GEMM B-tile staging a contiguous 16B-per-lane copy (async-friendly).
// ---------------------------------------------------------------------------
__global__ __launch_bounds__(256) void transpose_w_kernel(
    const float* __restrict__ src,         // one layer [D_][TWOD_]
    unsigned short* __restrict__ dst) {    // [TWOD_][K_]
  int i  = blockIdx.x * 256 + threadIdx.x; // over D_*TWOD_/4
  int k  = i / (TWOD_ / 4);
  int nb = (i % (TWOD_ / 4)) * 4;
  f32x4 v = *(const f32x4*)(src + (size_t)k * TWOD_ + nb);
#pragma unroll
  for (int j = 0; j < 4; ++j)
    dst[(size_t)(nb + j) * K_ + k] = f32_to_bf16_rne(v[j]);
}

// ---------------------------------------------------------------------------
// Windowed weighted sum with padding; emits f32 carry AND bf16 GEMM operand.
// ---------------------------------------------------------------------------
__global__ __launch_bounds__(256) void window_sum_kernel(
    const float* __restrict__ inputs, const float* __restrict__ lpad,
    const float* __restrict__ rpad,   const float* __restrict__ wts,
    float* __restrict__ Xf, unsigned short* __restrict__ Xb, int kbase) {
  int idx = blockIdx.x * 256 + threadIdx.x;        // over M_*D_/4
  int row = idx / (D_ / 4);
  int c4  = (idx % (D_ / 4)) * 4;
  int b = row / L_, l = row % L_;
  f32x4 acc = {0.f, 0.f, 0.f, 0.f};
#pragma unroll
  for (int t = 0; t <= W_; ++t) {
    int p = l + kbase + t;                          // padded row index
    const float* src;
    if (p < W_)             src = lpad   + (size_t)p * D_;
    else if (p < W_ + L_)   src = inputs + ((size_t)b * L_ + (p - W_)) * D_;
    else                    src = rpad   + (size_t)(p - W_ - L_) * D_;
    f32x4 v = *(const f32x4*)(src + c4);
    acc += v * wts[t];
  }
  size_t off = (size_t)row * D_ + c4;
  *(f32x4*)(Xf + off) = acc;
  u16x4 hb;
#pragma unroll
  for (int j = 0; j < 4; ++j) hb[j] = f32_to_bf16_rne(acc[j]);
  *(u16x4*)(Xb + off) = hb;
}

// ---------------------------------------------------------------------------
// Fused GEMM + highway layer, double-buffered LDS, async global->LDS staging.
//  proj = A @ W + bias; nl = proj[:,n]; gate = proj[:,n+D]; g = sigmoid(gate)
//  y = g*x + (1-g)*relu(nl)
// 256 threads = 8 waves (4 M x 2 N); wave tile 32x32 (nl) + 32x32 (gate).
// ---------------------------------------------------------------------------
__global__ __launch_bounds__(256) void gemm_highway_kernel(
    const unsigned short* __restrict__ Ab,  // [M_,512] bf16 (GEMM A)
    const float* __restrict__ Xc,           // [M_,512] f32 (highway carry x)
    const unsigned short* __restrict__ Wt,  // [1024][512] bf16, row = out col
    const float* __restrict__ bias,         // [1024]
    float* __restrict__ Hf,                 // layer out f32 (if !finalOut)
    unsigned short* __restrict__ Hb,        // layer out bf16 (if !finalOut)
    float* __restrict__ Fout,               // d_out (if finalOut)
    int finalOut, int colOff) {
  __shared__ unsigned short sA [2][BM * SA];
  __shared__ unsigned short sBn[2][BN * SB];
  __shared__ unsigned short sBg[2][BN * SB];

  const int tid   = threadIdx.x;
  const int lane  = tid & 31;
  const int wave  = tid >> 5;
  const int waveM = wave & 3;      // 32-row slab
  const int waveN = wave >> 2;     // 32-col slab
  const int half  = lane >> 4;
  const int l15   = lane & 15;

  const int n0 = blockIdx.x * BN;
  const int m0 = blockIdx.y * BM;

  // Stage one BK-slice into LDS buffer `buf` (pure 16B copies).
  auto stage = [&](int buf, int kt) {
    int c = tid;                                   // A: 512 chunks (2 rounds)
#pragma unroll
    for (int it = 0; it < 2; ++it, c += 256) {
      int r = c >> 2, cc = (c & 3) * 8;
      copy16_g2l(Ab + (size_t)(m0 + r) * K_ + kt + cc, &sA[buf][r * SA + cc]);
    }
    int nl = tid >> 2, cc2 = (tid & 3) * 8;        // B: 256 chunks each tile
    copy16_g2l(Wt + (size_t)(n0 + nl) * K_ + kt + cc2,
               &sBn[buf][nl * SB + cc2]);
    copy16_g2l(Wt + (size_t)(D_ + n0 + nl) * K_ + kt + cc2,
               &sBg[buf][nl * SB + cc2]);
  };

  v8f accN[2][2], accG[2][2];
#pragma unroll
  for (int mi = 0; mi < 2; ++mi)
#pragma unroll
    for (int ni = 0; ni < 2; ++ni) { accN[mi][ni] = {}; accG[mi][ni] = {}; }

  stage(0, 0);
  wait_async_copies();
  __syncthreads();

  int cur = 0;
  for (int kt = 0; kt < K_; kt += BK, cur ^= 1) {
    if (kt + BK < K_) stage(cur ^ 1, kt + BK);     // overlap with WMMA below

    // Fragments (k = 16*half + s in both A and B, not AMD's table order), contiguous b128 reads.
    union Frag { u32x4 q[2]; v16bf v; } a[2], bn[2], bg[2];
#pragma unroll
    for (int mi = 0; mi < 2; ++mi) {
      const unsigned int* base =
          (const unsigned int*)&sA[cur][(waveM * 32 + mi * 16 + l15) * SA];
      a[mi].q[0] = *(const u32x4*)(base + half * 8);
      a[mi].q[1] = *(const u32x4*)(base + half * 8 + 4);
    }
#pragma unroll
    for (int ni = 0; ni < 2; ++ni) {
      const unsigned int* pn =
          (const unsigned int*)&sBn[cur][(waveN * 32 + ni * 16 + l15) * SB];
      bn[ni].q[0] = *(const u32x4*)(pn + half * 8);
      bn[ni].q[1] = *(const u32x4*)(pn + half * 8 + 4);
      const unsigned int* pg =
          (const unsigned int*)&sBg[cur][(waveN * 32 + ni * 16 + l15) * SB];
      bg[ni].q[0] = *(const u32x4*)(pg + half * 8);
      bg[ni].q[1] = *(const u32x4*)(pg + half * 8 + 4);
    }

#pragma unroll
    for (int mi = 0; mi < 2; ++mi)
#pragma unroll
      for (int ni = 0; ni < 2; ++ni) {
        accN[mi][ni] = __builtin_amdgcn_wmma_f32_16x16x32_bf16(
            false, a[mi].v, false, bn[ni].v, (short)0, accN[mi][ni], false, false);
        accG[mi][ni] = __builtin_amdgcn_wmma_f32_16x16x32_bf16(
            false, a[mi].v, false, bg[ni].v, (short)0, accG[mi][ni], false, false);
      }

    wait_async_copies();   // next buffer written (own slice)
    __syncthreads();       // all slices visible; all reads of `cur` done
  }

  // Fused highway epilogue. C/D layout: VGPR r -> (M = r + 8*half, N = l15).
  const size_t OUTHALF = (size_t)M_ * TWOD_;
#pragma unroll
  for (int mi = 0; mi < 2; ++mi) {
#pragma unroll
    for (int ni = 0; ni < 2; ++ni) {
      int ng = n0 + waveN * 32 + ni * 16 + l15;
      float bN = bias[ng];
      float bG = bias[D_ + ng];
      int mBase = m0 + waveM * 32 + mi * 16 + half * 8;
#pragma unroll
      for (int r = 0; r < 8; ++r) {
        int m = mBase + r;
        float x  = Xc[(size_t)m * K_ + ng];
        float pn = accN[mi][ni][r] + bN;
        float pg = accG[mi][ni][r] + bG;
        float g  = 1.0f / (1.0f + __expf(-pg));
        float y  = g * x + (1.0f - g) * fmaxf(pn, 0.0f);
        if (finalOut) {
          size_t o = (size_t)m * TWOD_ + colOff + ng;
          Fout[o] = y;                               // all_layers[0]
          Fout[OUTHALF + o] = y;                     // last (duplicate)
        } else {
          Hf[(size_t)m * K_ + ng] = y;
          Hb[(size_t)m * K_ + ng] = f32_to_bf16_rne(y);
        }
      }
    }
  }
}

// ---------------------------------------------------------------------------
// Host launch
// ---------------------------------------------------------------------------
extern "C" void kernel_launch(void* const* d_in, const int* in_sizes, int n_in,
                              void* d_out, int out_size, void* d_ws, size_t ws_size,
                              hipStream_t stream) {
  const float* inputs = (const float*)d_in[0];
  const float* lpad   = (const float*)d_in[1];
  const float* rpad   = (const float*)d_in[2];
  const float* lw     = (const float*)d_in[3];
  const float* rw     = (const float*)d_in[4];
  const float* leftW  = (const float*)d_in[5];
  const float* leftB  = (const float*)d_in[6];
  const float* rightW = (const float*)d_in[7];
  const float* rightB = (const float*)d_in[8];
  float* out = (float*)d_out;

  // Workspace: Xf(64M) | Hf(64M) | Xb(32M) | Hb(32M) | Wt bf16 (4M)
  char* ws = (char*)d_ws;
  const size_t MD = (size_t)M_ * D_;
  float* Xf = (float*)ws;
  float* Hf = (float*)(ws + MD * 4);
  unsigned short* Xb = (unsigned short*)(ws + MD * 8);
  unsigned short* Hb = (unsigned short*)(ws + MD * 10);
  unsigned short* WtL = (unsigned short*)(ws + MD * 12);
  unsigned short* WtR = WtL + (size_t)NL_ * TWOD_ * K_;

  const int tBlocks = (D_ * TWOD_ / 4) / 256;     // 512 per layer
  for (int l = 0; l < NL_; ++l) {
    transpose_w_kernel<<<tBlocks, 256, 0, stream>>>(
        leftW + (size_t)l * D_ * TWOD_, WtL + (size_t)l * TWOD_ * K_);
    transpose_w_kernel<<<tBlocks, 256, 0, stream>>>(
        rightW + (size_t)l * D_ * TWOD_, WtR + (size_t)l * TWOD_ * K_);
  }

  const int winBlocks = (M_ * D_ / 4) / 256;      // 16384
  const dim3 gGrid(D_ / BN, M_ / BM);             // (8, 256)

  for (int side = 0; side < 2; ++side) {
    const float* taps        = side ? rw     : lw;
    const unsigned short* Wt = side ? WtR    : WtL;
    const float* bias        = side ? rightB : leftB;
    const int kbase          = side ? W_     : 0;
    const int colOff         = side ? D_     : 0;

    window_sum_kernel<<<winBlocks, 256, 0, stream>>>(inputs, lpad, rpad, taps,
                                                     Xf, Xb, kbase);
    // Layer 0: X -> H
    gemm_highway_kernel<<<gGrid, 256, 0, stream>>>(
        Xb, Xf, Wt, bias, Hf, Hb, out, 0, 0);
    // Layer 1: H -> d_out (both duplicated output halves)
    gemm_highway_kernel<<<gGrid, 256, 0, stream>>>(
        Hb, Hf, Wt + (size_t)TWOD_ * K_, bias + TWOD_, nullptr, nullptr, out,
        1, colOff);
  }
}
